// GPSDMPNNEncoder_42219528519695
// MI455X (gfx1250) — compile-verified, matrix operand order checked by reading, not run on hardware
//
#include <hip/hip_runtime.h>
#include <math.h>

typedef __attribute__((ext_vector_type(16))) __bf16 v16bf;
typedef __attribute__((ext_vector_type(8)))  float  v8f;

union Frag { uint4 q[2]; v16bf v; };

__device__ __forceinline__ unsigned short f2bf(float f) {
  unsigned int u = __float_as_uint(f);
  u += 0x7FFFu + ((u >> 16) & 1u);          // round-to-nearest-even
  return (unsigned short)(u >> 16);
}
__device__ __forceinline__ float gelu_erf(float x) {
  return 0.5f * x * (1.0f + erff(x * 0.70710678118654752f));
}

// ---------------------------------------------------------------------------
// bf16 WMMA GEMM: out = [residual +] act( A[M,K] @ B[K,N] + bias )
//   A: row-major bf16 [M, lda]   (lda multiple of 8, K padded to mult. of 32)
//   Bt: TRANSPOSED weights bf16 [N, ldb]  (ldb = Kpad)
// Block tile 128x128, 256 thr = 8 waves (4 along M x 2 along N),
// each wave: 32x64 = 2x4 wmma accumulators. Double-buffered LDS, branchless
// inner loop, all staging traffic is b128.
// ---------------------------------------------------------------------------
#define TM 128
#define TN 128
#define TK 32
#define LP 40   // LDS pitch in halves (80B rows keep every frag load 16B-aligned)

__global__ __launch_bounds__(256)
void gemm_bf16_wmma(const unsigned short* __restrict__ A,
                    const unsigned short* __restrict__ Bt,
                    const float* __restrict__ bias,
                    const float* __restrict__ residual,
                    float* __restrict__ outF,
                    unsigned short* __restrict__ outB,
                    int M, int N, int K, int lda, int ldb, int act)
{
  __shared__ __align__(16) unsigned short As[2][TM][LP];
  __shared__ __align__(16) unsigned short Bs[2][TN][LP];

  const int tid  = threadIdx.x;
  const int lane = tid & 31;
  const int wave = tid >> 5;
  const int wm   = wave & 3;          // 4 waves along M (32 rows each)
  const int wn   = wave >> 2;         // 2 waves along N (64 cols each)
  const int m0   = blockIdx.y * TM;
  const int n0   = blockIdx.x * TN;
  const int half = lane >> 4;
  const int l15  = lane & 15;

  // fixed per-thread staging slots: rows r and r+64, 8-half chunk c
  const int r = tid >> 2;
  const int c = tid & 3;
  const unsigned short* pa = A  + (size_t)(m0 + r) * lda + c * 8;
  const unsigned short* pb = Bt + (size_t)(n0 + r) * ldb + c * 8;
  const size_t a64 = (size_t)64 * lda;
  const size_t b64 = (size_t)64 * ldb;

  v8f acc[2][4] = {};

  // prologue: stage tile 0 into buffer 0
  uint4 ra0 = *reinterpret_cast<const uint4*>(pa);
  uint4 ra1 = *reinterpret_cast<const uint4*>(pa + a64);
  uint4 rb0 = *reinterpret_cast<const uint4*>(pb);
  uint4 rb1 = *reinterpret_cast<const uint4*>(pb + b64);
  *reinterpret_cast<uint4*>(&As[0][r][c * 8])      = ra0;
  *reinterpret_cast<uint4*>(&As[0][r + 64][c * 8]) = ra1;
  *reinterpret_cast<uint4*>(&Bs[0][r][c * 8])      = rb0;
  *reinterpret_cast<uint4*>(&Bs[0][r + 64][c * 8]) = rb1;

  int cur = 0;
  for (int kb = 0; kb < K; kb += TK) {
    // branchless next-tile k (redundant reload of current tile on last iter)
    int kn = (kb + TK < K) ? (kb + TK) : kb;
    int kp = (kn + TK < K) ? (kn + TK) : kn;
    __builtin_prefetch(pa + kp, 0, 1);                    // global_prefetch_b8
    ra0 = *reinterpret_cast<const uint4*>(pa + kn);
    ra1 = *reinterpret_cast<const uint4*>(pa + a64 + kn);
    rb0 = *reinterpret_cast<const uint4*>(pb + kn);
    rb1 = *reinterpret_cast<const uint4*>(pb + b64 + kn);

    __syncthreads();   // stores of current tile visible

    const unsigned short (*cA)[LP] = As[cur];
    const unsigned short (*cB)[LP] = Bs[cur];
    Frag a[2], b[4];
#pragma unroll
    for (int mi = 0; mi < 2; ++mi) {
      a[mi].q[0] = *reinterpret_cast<const uint4*>(&cA[wm * 32 + mi * 16 + l15][8 * half]);
      a[mi].q[1] = *reinterpret_cast<const uint4*>(&cA[wm * 32 + mi * 16 + l15][16 + 8 * half]);
    }
#pragma unroll
    for (int ni = 0; ni < 4; ++ni) {
      b[ni].q[0] = *reinterpret_cast<const uint4*>(&cB[wn * 64 + ni * 16 + l15][8 * half]);
      b[ni].q[1] = *reinterpret_cast<const uint4*>(&cB[wn * 64 + ni * 16 + l15][16 + 8 * half]);
    }
#pragma unroll
    for (int mi = 0; mi < 2; ++mi)
#pragma unroll
      for (int ni = 0; ni < 4; ++ni)
        acc[mi][ni] = __builtin_amdgcn_wmma_f32_16x16x32_bf16(
            false, a[mi].v, false, b[ni].v, (short)0, acc[mi][ni], false, false);

    __syncthreads();   // everyone done reading buf cur^1 (previous iteration)

    int nxt = cur ^ 1;
    *reinterpret_cast<uint4*>(&As[nxt][r][c * 8])      = ra0;
    *reinterpret_cast<uint4*>(&As[nxt][r + 64][c * 8]) = ra1;
    *reinterpret_cast<uint4*>(&Bs[nxt][r][c * 8])      = rb0;
    *reinterpret_cast<uint4*>(&Bs[nxt][r + 64][c * 8]) = rb1;
    cur = nxt;
  }

  // ---- epilogue: C layout => VGPR rr holds (M = rr + 8*half, N = l15)
#pragma unroll
  for (int mi = 0; mi < 2; ++mi)
#pragma unroll
    for (int ni = 0; ni < 4; ++ni)
#pragma unroll
      for (int rr = 0; rr < 8; ++rr) {
        int gm = m0 + wm * 32 + mi * 16 + rr + 8 * half;
        int gn = n0 + wn * 64 + ni * 16 + l15;
        float v = acc[mi][ni][rr];
        if (bias)     v += bias[gn];
        if (act)      v  = gelu_erf(v);
        size_t o = (size_t)gm * N + gn;
        if (residual) v += residual[o];
        if (outF) outF[o] = v;
        if (outB) outB[o] = f2bf(v);
      }
}

// ---------------------------------------------------------------------------
// Weight convert f32 [K,N] -> TRANSPOSED bf16 [N, Kpad] with zero K padding
// ---------------------------------------------------------------------------
__global__ void cvt_padT_bf16(const float* __restrict__ src, unsigned short* __restrict__ dst,
                              int K, int N, int Kpad) {
  int idx = blockIdx.x * 256 + threadIdx.x;
  int total = N * Kpad;
  if (idx >= total) return;
  int n  = idx / Kpad;
  int kk = idx - n * Kpad;
  float v = (kk < K) ? src[(size_t)kk * N + n] : 0.0f;
  dst[idx] = f2bf(v);
}

// plain f32 -> bf16 convert
__global__ void cvt_bf16(const float* __restrict__ src, unsigned short* __restrict__ dst,
                         int total) {
  int idx = blockIdx.x * 256 + threadIdx.x;
  if (idx >= total) return;
  dst[idx] = f2bf(src[idx]);
}

// concat(f_atoms[b2a[b]], f_bonds[b]) zero-padded to 192 cols, bf16
__global__ void build_bond_in(const float* __restrict__ fa, const float* __restrict__ fb,
                              const int* __restrict__ b2a, unsigned short* __restrict__ dst,
                              int NB) {
  int idx = blockIdx.x * 256 + threadIdx.x;
  if (idx >= NB * 192) return;
  int b = idx / 192;
  int c = idx - b * 192;
  float v = 0.0f;
  if (c < 153)      v = fa[(size_t)b2a[b] * 153 + c];
  else if (c < 167) v = fb[(size_t)b * 14 + (c - 153)];
  dst[idx] = f2bf(v);
}

// concat(f_atoms, a_message) zero-padded to 416 cols, bf16
__global__ void build_atom_in(const float* __restrict__ fa, const float* __restrict__ am,
                              unsigned short* __restrict__ dst, int NA) {
  int idx = blockIdx.x * 256 + threadIdx.x;
  if (idx >= NA * 416) return;
  int a = idx / 416;
  int c = idx - a * 416;
  float v = 0.0f;
  if (c < 153)      v = fa[(size_t)a * 153 + c];
  else if (c < 409) v = am[(size_t)a * 256 + (c - 153)];
  dst[idx] = f2bf(v);
}

// nei[a,c] = sum_{j<6} msg[a2b[a,j], c]
__global__ void gather_sum6(const float* __restrict__ msg, const int* __restrict__ a2b,
                            float* __restrict__ nei, int NA) {
  int idx = blockIdx.x * 256 + threadIdx.x;
  if (idx >= NA * 256) return;
  int a = idx >> 8;
  int c = idx & 255;
  float s = 0.0f;
#pragma unroll
  for (int j = 0; j < 6; ++j) s += msg[(size_t)a2b[a * 6 + j] * 256 + c];
  nei[idx] = s;
}

// one wave per bond row: LN(nei[b2a[b]] - msg[b2revb[b]]) -> bf16
__global__ __launch_bounds__(256)
void bond_ln_bf16(const float* __restrict__ nei, const float* __restrict__ msg,
                  const int* __restrict__ b2a, const int* __restrict__ b2revb,
                  const float* __restrict__ sc, const float* __restrict__ sh,
                  unsigned short* __restrict__ dst) {
  int lane = threadIdx.x & 31;
  int wave = threadIdx.x >> 5;
  int row  = blockIdx.x * 8 + wave;
  size_t na = (size_t)b2a[row] * 256;
  size_t nr = (size_t)b2revb[row] * 256;
  float v[8]; float sum = 0.0f;
#pragma unroll
  for (int i = 0; i < 8; ++i) { int c = i * 32 + lane; v[i] = nei[na + c] - msg[nr + c]; sum += v[i]; }
#pragma unroll
  for (int off = 16; off > 0; off >>= 1) sum += __shfl_xor(sum, off, 32);
  float mean = sum * (1.0f / 256.0f);
  float var = 0.0f;
#pragma unroll
  for (int i = 0; i < 8; ++i) { float d = v[i] - mean; var += d * d; }
#pragma unroll
  for (int off = 16; off > 0; off >>= 1) var += __shfl_xor(var, off, 32);
  float inv = rsqrtf(var * (1.0f / 256.0f) + 1e-5f);
#pragma unroll
  for (int i = 0; i < 8; ++i) {
    int c = i * 32 + lane;
    dst[(size_t)row * 256 + c] = f2bf((v[i] - mean) * inv * sc[c] + sh[c]);
  }
}

// one wave per row LN: optional f32 and/or bf16 outputs
__global__ __launch_bounds__(256)
void ln_rows(const float* __restrict__ in, const float* __restrict__ sc,
             const float* __restrict__ sh, float* __restrict__ outF,
             unsigned short* __restrict__ outB) {
  int lane = threadIdx.x & 31;
  int wave = threadIdx.x >> 5;
  size_t row = (size_t)blockIdx.x * 8 + wave;
  const float* rp = in + row * 256;
  float v[8]; float sum = 0.0f;
#pragma unroll
  for (int i = 0; i < 8; ++i) { v[i] = rp[i * 32 + lane]; sum += v[i]; }
#pragma unroll
  for (int off = 16; off > 0; off >>= 1) sum += __shfl_xor(sum, off, 32);
  float mean = sum * (1.0f / 256.0f);
  float var = 0.0f;
#pragma unroll
  for (int i = 0; i < 8; ++i) { float d = v[i] - mean; var += d * d; }
#pragma unroll
  for (int off = 16; off > 0; off >>= 1) var += __shfl_xor(var, off, 32);
  float inv = rsqrtf(var * (1.0f / 256.0f) + 1e-5f);
#pragma unroll
  for (int i = 0; i < 8; ++i) {
    int c = i * 32 + lane;
    float o = (v[i] - mean) * inv * sc[c] + sh[c];
    if (outF) outF[row * 256 + c] = o;
    if (outB) outB[row * 256 + c] = f2bf(o);
  }
}

// per-(mol, head) softmax attention over 48 atoms, head dim 64; ctx -> bf16
__global__ __launch_bounds__(64)
void attn_kernel(const float* __restrict__ q, const float* __restrict__ k,
                 const float* __restrict__ v, unsigned short* __restrict__ ctx) {
  int bh = blockIdx.x;
  int b = bh >> 2, h = bh & 3;
  __shared__ float qs[48][64], ks[48][64], vs[48][64];
  int t = threadIdx.x;
#pragma unroll 4
  for (int i = 0; i < 48; ++i) {
    int lin = i * 64 + t;
    int si = lin >> 6, d = lin & 63;
    size_t base = ((size_t)(b * 48 + si)) * 256 + h * 64 + d;
    qs[si][d] = q[base]; ks[si][d] = k[base]; vs[si][d] = v[base];
  }
  __syncthreads();
  if (t < 48) {
    float sc[48], mx = -1e30f;
    for (int kk = 0; kk < 48; ++kk) {
      float a = 0.0f;
#pragma unroll 8
      for (int d = 0; d < 64; ++d) a += qs[t][d] * ks[kk][d];
      a *= 0.125f;              // 1/sqrt(64)
      sc[kk] = a; mx = fmaxf(mx, a);
    }
    float den = 0.0f;
    for (int kk = 0; kk < 48; ++kk) { sc[kk] = __expf(sc[kk] - mx); den += sc[kk]; }
    float rden = 1.0f / den;
    for (int d = 0; d < 64; ++d) {
      float a = 0.0f;
#pragma unroll 8
      for (int kk = 0; kk < 48; ++kk) a += sc[kk] * vs[kk][d];
      ctx[((size_t)(b * 48 + t)) * 256 + h * 64 + d] = f2bf(a * rden);
    }
  }
}

// per-mol attention-weighted readout
__global__ __launch_bounds__(64)
void readout_kernel(const float* __restrict__ keys, const float* __restrict__ x,
                    const float* __restrict__ query, float* __restrict__ out) {
  int b = blockIdx.x, t = threadIdx.x;
  __shared__ float sc[48];
  if (t < 48) {
    float a = 0.0f;
    const float* kr = keys + ((size_t)(b * 48 + t)) * 256;
#pragma unroll 8
    for (int c = 0; c < 256; ++c) a += kr[c] * query[c];
    sc[t] = a;
  }
  __syncthreads();
  float mx = -1e30f;
  for (int s = 0; s < 48; ++s) mx = fmaxf(mx, sc[s]);
  float den = 0.0f;
  for (int s = 0; s < 48; ++s) den += __expf(sc[s] - mx);
  float rden = 1.0f / den;
  for (int cc = 0; cc < 4; ++cc) {
    int c = t + cc * 64;
    float a = 0.0f;
    for (int s = 0; s < 48; ++s)
      a += __expf(sc[s] - mx) * rden * x[((size_t)(b * 48 + s)) * 256 + c];
    out[(size_t)b * 256 + c] = a;
  }
}

// ---------------------------------------------------------------------------
extern "C" void kernel_launch(void* const* d_in, const int* in_sizes, int n_in,
                              void* d_out, int out_size, void* d_ws, size_t ws_size,
                              hipStream_t stream) {
  const float* f_atoms   = (const float*)d_in[0];
  const float* f_bonds   = (const float*)d_in[1];
  const int*   a2b       = (const int*)d_in[2];
  const int*   b2a       = (const int*)d_in[3];
  const int*   b2revb    = (const int*)d_in[4];
  const float* W_i       = (const float*)d_in[5];
  const float* b_i       = (const float*)d_in[6];
  const float* W_h       = (const float*)d_in[7];
  const float* b_h       = (const float*)d_in[8];
  const float* ln_msg_s  = (const float*)d_in[9];
  const float* ln_msg_b  = (const float*)d_in[10];
  const float* W_o       = (const float*)d_in[11];
  const float* b_o       = (const float*)d_in[12];
  const float* ln_atom_s = (const float*)d_in[13];
  const float* ln_atom_b = (const float*)d_in[14];
  const float* q_w = (const float*)d_in[15]; const float* q_b = (const float*)d_in[16];
  const float* k_w = (const float*)d_in[17]; const float* k_b = (const float*)d_in[18];
  const float* v_w = (const float*)d_in[19]; const float* v_b = (const float*)d_in[20];
  const float* o_w = (const float*)d_in[21]; const float* o_b = (const float*)d_in[22];
  const float* ln1_s = (const float*)d_in[23]; const float* ln1_b = (const float*)d_in[24];
  const float* ln2_s = (const float*)d_in[25]; const float* ln2_b = (const float*)d_in[26];
  const float* ff1_w = (const float*)d_in[27]; const float* ff1_b = (const float*)d_in[28];
  const float* ff2_w = (const float*)d_in[29]; const float* ff2_b = (const float*)d_in[30];
  const float* rq    = (const float*)d_in[31];
  const float* rk_w  = (const float*)d_in[32]; const float* rk_b = (const float*)d_in[33];

  const int NA = in_sizes[0] / 153;      // 49152 atoms
  const int NB = in_sizes[3];            // 147456 directed bonds
  const int n_mols = NA / 48;            // uniform scopes, apm = 48
  const int H = 256;

  // ---- workspace carve-up
  char* wp = (char*)d_ws;
  auto carve = [&](size_t bytes) { void* p = wp; wp += (bytes + 255) & ~(size_t)255; return p; };
  float*          msg    = (float*)carve((size_t)NB * H * 4);
  float*          nei    = (float*)carve((size_t)NA * H * 4);
  unsigned short* stageA = (unsigned short*)carve((size_t)NB * H * 2); // bond-in(192)/ln-out(256)/atom-in(416)
  unsigned short* wsb    = (unsigned short*)carve((size_t)942080 * 2);
  float*          x      = (float*)carve((size_t)NA * H * 4);
  unsigned short* xb     = (unsigned short*)carve((size_t)NA * H * 2);
  float*          qf     = (float*)carve((size_t)NA * H * 4);          // later reused for keys
  float*          kf     = (float*)carve((size_t)NA * H * 4);
  float*          vf     = (float*)carve((size_t)NA * H * 4);
  unsigned short* ctxb   = (unsigned short*)carve((size_t)NA * H * 2);
  unsigned short* ffmid  = (unsigned short*)carve((size_t)NA * 512 * 2);

  // transposed bf16 weights [N][Kpad]
  unsigned short* Wib = wsb;                    // 256 x 192
  unsigned short* Whb = Wib + 256 * 192;        // 3 x (256 x 256)
  unsigned short* Wob = Whb + 3 * 65536;        // 256 x 416
  unsigned short* Qw  = Wob + 256 * 416;
  unsigned short* Kw  = Qw + 65536;
  unsigned short* Vw  = Kw + 65536;
  unsigned short* Ow  = Vw + 65536;
  unsigned short* F1w = Ow + 65536;             // 512 x 256
  unsigned short* F2w = F1w + 512 * 256;        // 256 x 512
  unsigned short* Rkw = F2w + 256 * 512;

  auto cvtT = [&](const float* s, unsigned short* d, int K, int N, int Kp) {
    int tot = N * Kp;
    cvt_padT_bf16<<<(tot + 255) / 256, 256, 0, stream>>>(s, d, K, N, Kp);
  };
  auto gemm = [&](const unsigned short* A, const unsigned short* B, const float* bias,
                  const float* res, float* oF, unsigned short* oB,
                  int M, int N, int K, int lda, int act) {
    gemm_bf16_wmma<<<dim3(N / 128, M / 128), 256, 0, stream>>>(
        A, B, bias, res, oF, oB, M, N, K, lda, /*ldb=*/K, act);
  };

  // ---- weights to transposed bf16 (K zero-padded to multiples of 32)
  cvtT(W_i, Wib, 167, 256, 192);
  for (int t = 0; t < 3; ++t) cvtT(W_h + (size_t)t * 65536, Whb + (size_t)t * 65536, 256, 256, 256);
  cvtT(W_o, Wob, 409, 256, 416);
  cvtT(q_w, Qw, 256, 256, 256);  cvtT(k_w, Kw, 256, 256, 256);
  cvtT(v_w, Vw, 256, 256, 256);  cvtT(o_w, Ow, 256, 256, 256);
  cvtT(ff1_w, F1w, 256, 512, 256);
  cvtT(ff2_w, F2w, 512, 256, 512);
  cvtT(rk_w, Rkw, 256, 256, 256);

  // ---- D-MPNN: initial bond projection
  build_bond_in<<<(NB * 192 + 255) / 256, 256, 0, stream>>>(f_atoms, f_bonds, b2a, stageA, NB);
  gemm(stageA, Wib, b_i, nullptr, msg, nullptr, NB, 256, 192, 192, 1);

  // ---- 3 message-passing iterations
  for (int t = 0; t < 3; ++t) {
    gather_sum6<<<NA, 256, 0, stream>>>(msg, a2b, nei, NA);
    bond_ln_bf16<<<NB / 8, 256, 0, stream>>>(nei, msg, b2a, b2revb,
                                             ln_msg_s + t * 256, ln_msg_b + t * 256, stageA);
    gemm(stageA, Whb + (size_t)t * 65536, b_h + t * 256, msg, msg, nullptr, NB, 256, 256, 256, 1);
  }

  // ---- atom readback + output projection + atom LN
  gather_sum6<<<NA, 256, 0, stream>>>(msg, a2b, nei, NA);
  build_atom_in<<<(NA * 416 + 255) / 256, 256, 0, stream>>>(f_atoms, nei, stageA, NA);
  gemm(stageA, Wob, b_o, nullptr, x, nullptr, NA, 256, 416, 416, 1);
  ln_rows<<<NA / 8, 256, 0, stream>>>(x, ln_atom_s, ln_atom_b, x, nullptr);

  // ---- transformer encoder layer (pre-norm)
  ln_rows<<<NA / 8, 256, 0, stream>>>(x, ln1_s, ln1_b, nullptr, xb);
  gemm(xb, Qw, q_b, nullptr, qf, nullptr, NA, 256, 256, 256, 0);
  gemm(xb, Kw, k_b, nullptr, kf, nullptr, NA, 256, 256, 256, 0);
  gemm(xb, Vw, v_b, nullptr, vf, nullptr, NA, 256, 256, 256, 0);
  attn_kernel<<<n_mols * 4, 64, 0, stream>>>(qf, kf, vf, ctxb);
  gemm(ctxb, Ow, o_b, x, x, nullptr, NA, 256, 256, 256, 0);
  ln_rows<<<NA / 8, 256, 0, stream>>>(x, ln2_s, ln2_b, nullptr, xb);
  gemm(xb, F1w, ff1_b, nullptr, nullptr, ffmid, NA, 512, 256, 256, 1);
  gemm(ffmid, F2w, ff2_b, x, x, nullptr, NA, 256, 512, 512, 0);

  // ---- attention-weighted readout
  cvt_bf16<<<(NA * 256 + 255) / 256, 256, 0, stream>>>(x, xb, NA * 256);
  gemm(xb, Rkw, rk_b, nullptr, qf /*keys*/, nullptr, NA, 256, 256, 256, 0);
  readout_kernel<<<n_mols, 64, 0, stream>>>(qf, x, rq, (float*)d_out);

  (void)n_in; (void)out_size; (void)ws_size;
}
